// GLGAT_91036126806477
// MI455X (gfx1250) — matrix operand order repaired and checked by reading, not compiled, not run on hardware
//
#include <hip/hip_runtime.h>
#include <hip/hip_bf16.h>

// ---------------------------------------------------------------------------
// Types for CDNA5 WMMA (wave32, 16x16x32 bf16 -> f32)
// ---------------------------------------------------------------------------
typedef __attribute__((ext_vector_type(8)))  float  v8f;
typedef __attribute__((ext_vector_type(8)))  __bf16 v8bf;
typedef __attribute__((ext_vector_type(16))) __bf16 v16bf;

union BF16x16 { v16bf v; v8bf h[2]; };

#define GAT_ALPHA 0.2f
#define GAT_NEG   (-9.0e15f)

__device__ __forceinline__ v8f wmma_bf16(v16bf a, v16bf b, v8f c) {
    return __builtin_amdgcn_wmma_f32_16x16x32_bf16(
        /*neg_a=*/false, a, /*neg_b=*/false, b,
        /*c_mod=*/(short)0, c, /*reuse_a=*/false, /*reuse_b=*/false);
}

__device__ __forceinline__ v8bf pack8(float4 a, float4 b) {
    v8bf r;
    r[0] = (__bf16)a.x; r[1] = (__bf16)a.y; r[2] = (__bf16)a.z; r[3] = (__bf16)a.w;
    r[4] = (__bf16)b.x; r[5] = (__bf16)b.y; r[6] = (__bf16)b.z; r[7] = (__bf16)b.w;
    return r;
}

// ---------------------------------------------------------------------------
// Generic tiled GEMM: C[M,Nc] = act(A[M,K] @ B(K x Nc, row stride ldb) + bias)
// A,B f32 in memory, converted to bf16 for WMMA; accumulate f32.
// Block: 256 threads (8 waves). Block tile: 128(M) x 64(N), K-step 32.
// Wave grid 4x2; each wave computes 2x2 16x16 WMMA tiles.
// act: 0=none, 1=relu
// ---------------------------------------------------------------------------
__global__ __launch_bounds__(256)
void gemm_bf16_wmma(const float* __restrict__ A, const float* __restrict__ Bw,
                    const float* __restrict__ bias, float* __restrict__ C,
                    int M, int K, int Nc, int ldb, int act) {
    __shared__ __bf16 sA[128 * 40];   // row-major [row][k], stride 40 halfs (80B, 16B aligned)
    __shared__ __bf16 sB[64 * 40];    // transposed  [col][k], stride 40 halfs

    const int tid  = threadIdx.x;
    const int wave = tid >> 5;
    const int lane = tid & 31;
    const int waveM = wave & 3;       // 0..3 -> 32 rows each
    const int waveN = wave >> 2;      // 0..1 -> 32 cols each

    const int rowBase = blockIdx.y * 128;
    const int colBase = blockIdx.x * 64;

    v8f acc[2][2];
#pragma unroll
    for (int i = 0; i < 2; ++i)
#pragma unroll
        for (int j = 0; j < 2; ++j) acc[i][j] = (v8f)0.0f;

    const int mrow = lane & 15;
    const int aBase = (lane < 16) ? 0 : 8;   // A fragment K sub-base
    const int kB    = (lane < 16) ? 0 : 16;  // unused; B fragments use aBase
    const int ncol  = lane & 15;

    // staging indices (constant over K loop)
    const int stRow = tid >> 1;          // 0..127
    const int stSeg = tid & 1;           // 0..1
    const int stGr  = rowBase + stRow;
    const int stN   = tid & 63;          // B stage col
    const int stGn  = colBase + stN;
    const int stKk  = tid >> 6;          // 0..3

    for (int k0 = 0; k0 < K; k0 += 32) {
        __syncthreads();
        // ---- stage A tile: 128 rows x 32 k (2 threads per row, 16 f32 each)
        {
            v8bf* dst = (v8bf*)&sA[stRow * 40 + stSeg * 16];   // 16B aligned
            if (stGr < M) {
                const float* src = A + (size_t)stGr * K + k0 + stSeg * 16;
                float4 f0 = *(const float4*)(src + 0);
                float4 f1 = *(const float4*)(src + 4);
                float4 f2 = *(const float4*)(src + 8);
                float4 f3 = *(const float4*)(src + 12);
                dst[0] = pack8(f0, f1);
                dst[1] = pack8(f2, f3);
                if (k0 + 32 < K) __builtin_prefetch(src + 32, 0, 1);  // next K-chunk
            } else {
                v8bf z = {};
                dst[0] = z;
                dst[1] = z;
            }
        }
        // ---- stage B tile transposed: sB[col][k] = B[(k0+k)*ldb + colBase+col]
        {
#pragma unroll
            for (int i = 0; i < 8; ++i) {
                const int k = stKk + i * 4;
                float v = (stGn < Nc) ? Bw[(size_t)(k0 + k) * ldb + stGn] : 0.0f;
                sB[stN * 40 + k] = (__bf16)v;
            }
        }
        __syncthreads();

        // ---- fragments + 4 WMMAs per wave
        BF16x16 afr[2], bfr[2];
#pragma unroll
        for (int mi = 0; mi < 2; ++mi) {
            const int row = waveM * 32 + mi * 16 + mrow;
            afr[mi].h[0] = *(const v8bf*)&sA[row * 40 + aBase];        // K = aBase..aBase+7
            afr[mi].h[1] = *(const v8bf*)&sA[row * 40 + 16 + aBase];   // K = 16+aBase..
        }
#pragma unroll
        for (int ni = 0; ni < 2; ++ni) {
            const int col = waveN * 32 + ni * 16 + ncol;
            bfr[ni].h[0] = *(const v8bf*)&sB[col * 40 + aBase];        // K = aBase..aBase+7
            bfr[ni].h[1] = *(const v8bf*)&sB[col * 40 + 16 + aBase];   // K = 16+aBase..
        }
#pragma unroll
        for (int mi = 0; mi < 2; ++mi)
#pragma unroll
            for (int ni = 0; ni < 2; ++ni)
                acc[mi][ni] = wmma_bf16(afr[mi].v, bfr[ni].v, acc[mi][ni]);
    }

    // ---- epilogue: C/D layout -> lane<16: M=r, N=lane ; lane>=16: M=r+8, N=lane-16
#pragma unroll
    for (int mi = 0; mi < 2; ++mi)
#pragma unroll
        for (int ni = 0; ni < 2; ++ni) {
            const int gn = colBase + waveN * 32 + ni * 16 + ncol;
#pragma unroll
            for (int r = 0; r < 8; ++r) {
                const int mloc = (lane < 16) ? r : (r + 8);
                const int gm = rowBase + waveM * 32 + mi * 16 + mloc;
                if (gm < M && gn < Nc) {
                    float v = acc[mi][ni][r];
                    if (bias) v += bias[gn];
                    if (act == 1) v = v > 0.0f ? v : 0.0f;
                    C[(size_t)gm * Nc + gn] = v;
                }
            }
        }
}

// ---------------------------------------------------------------------------
// Score projections: s1[i] = Wh[i,:] . a[0:H],  s2[i] = Wh[i,:] . a[H:2H]
// ---------------------------------------------------------------------------
__global__ __launch_bounds__(256)
void score_proj(const float* __restrict__ Wh, const float* __restrict__ a,
                float* __restrict__ s1, float* __restrict__ s2, int nNodes, int H) {
    int i = blockIdx.x * blockDim.x + threadIdx.x;
    if (i >= nNodes) return;
    const float* w = Wh + (size_t)i * H;
    float acc1 = 0.0f, acc2 = 0.0f;
    for (int h = 0; h < H; ++h) {
        float v = w[h];
        acc1 += v * a[h];
        acc2 += v * a[H + h];
    }
    s1[i] = acc1;
    s2[i] = acc2;
}

// ---------------------------------------------------------------------------
// Fused GAT attention (flash-style online softmax + WMMA for P @ Wh).
// out[b,i,:] = (elu?)( softmax_j( mask(leaky(s1_i+s2_j), adj) ) @ Wh[b,:, :] )
// One wave handles one 16-row i-tile. Block = 4 waves = 64 rows (same b).
// Wh j-chunk (32 rows) staged in LDS transposed [col][k] for B-fragments.
// NT = number of 16-wide N tiles covering H (H<=NT*16).
// ---------------------------------------------------------------------------
template <int NT>
__global__ __launch_bounds__(128)
void gat_attention(const float* __restrict__ Wh, const float* __restrict__ adj,
                   const float* __restrict__ s1, const float* __restrict__ s2,
                   float* __restrict__ out, int Nn, int H, int doElu) {
    __shared__ __bf16 sWhT[NT * 16 * 40];   // [col][k], stride 40 halfs

    const int tilesPerB = Nn / 64;
    const int b      = blockIdx.x / tilesPerB;
    const int i_base = (blockIdx.x % tilesPerB) * 64;
    const int wave   = threadIdx.x >> 5;
    const int lane   = threadIdx.x & 31;
    const int i0     = i_base + wave * 16;

    const int mrow = lane & 15;
    const int base = (lane < 16) ? 0 : 8;   // A-fragment K sub-base
    const int kB   = (lane < 16) ? 0 : 16;  // unused; B fragments use base
    const int ncol = lane & 15;

    const int grow = i0 + mrow;
    const float s1row = s1[b * Nn + grow];
    const float* adjRowBase = adj + ((size_t)b * Nn + grow) * Nn;

    float m = GAT_NEG;   // running row max (matches ref: all-masked row -> uniform)
    float l = 0.0f;      // running row denom
    v8f acc[NT];
#pragma unroll
    for (int nt = 0; nt < NT; ++nt) acc[nt] = (v8f)0.0f;

    for (int j0 = 0; j0 < Nn; j0 += 32) {
        __syncthreads();
        // ---- stage Wh chunk transposed: sWhT[c][k] = Wh[b, j0+k, c]
        if constexpr (NT == 8) {
            // each thread owns one full LDS column: 32 contiguous halfs -> 4x b128
            const int c = threadIdx.x;           // 0..127 == H
            float f[32];
#pragma unroll
            for (int k = 0; k < 32; ++k)
                f[k] = Wh[((size_t)(b * Nn + j0 + k)) * H + c];
            v8bf* dst = (v8bf*)&sWhT[c * 40];    // c*80 bytes, 16B aligned
#pragma unroll
            for (int q = 0; q < 4; ++q) {
                v8bf p;
#pragma unroll
                for (int e = 0; e < 8; ++e) p[e] = (__bf16)f[q * 8 + e];
                dst[q] = p;
            }
        } else {
            const int Cc = NT * 16;
            const int c  = threadIdx.x % Cc;
            const int k0 = threadIdx.x / Cc;
            const int ks = 128 / Cc;
            for (int k = k0; k < 32; k += ks) {
                float v = (c < H) ? Wh[((size_t)(b * Nn + j0 + k)) * H + c] : 0.0f;
                sWhT[c * 40 + k] = (__bf16)v;
            }
        }
        __syncthreads();

        // ---- masked scores for this lane's (row, 16 j-positions)
        const float* adjrow = adjRowBase + j0;
        const float* s2p    = s2 + b * Nn + j0;
        float4 A0 = *(const float4*)(adjrow + base);
        float4 A1 = *(const float4*)(adjrow + base + 4);
        float4 A2 = *(const float4*)(adjrow + 16 + base);
        float4 A3 = *(const float4*)(adjrow + 16 + base + 4);
        if (j0 + 32 < Nn) __builtin_prefetch(adjrow + 32 + base, 0, 1);  // next adj chunk
        float4 S0 = *(const float4*)(s2p + base);
        float4 S1 = *(const float4*)(s2p + base + 4);
        float4 S2 = *(const float4*)(s2p + 16 + base);
        float4 S3 = *(const float4*)(s2p + 16 + base + 4);
        float av[16] = {A0.x,A0.y,A0.z,A0.w, A1.x,A1.y,A1.z,A1.w,
                        A2.x,A2.y,A2.z,A2.w, A3.x,A3.y,A3.z,A3.w};
        float sv[16] = {S0.x,S0.y,S0.z,S0.w, S1.x,S1.y,S1.z,S1.w,
                        S2.x,S2.y,S2.z,S2.w, S3.x,S3.y,S3.z,S3.w};

        float ev[16];
        float lmax = GAT_NEG;
#pragma unroll
        for (int t = 0; t < 16; ++t) {
            float e = s1row + sv[t];
            e = e > 0.0f ? e : GAT_ALPHA * e;            // leaky_relu
            e = (av[t] > 0.0f) ? e : GAT_NEG;            // adjacency mask
            ev[t] = e;
            lmax = fmaxf(lmax, e);
        }
        // combine with partner lane (same row, other half of the 32 j's)
        float cmax = fmaxf(lmax, __shfl_xor(lmax, 16));
        float mnew = fmaxf(m, cmax);
        float psum = 0.0f;
        float p[16];
#pragma unroll
        for (int t = 0; t < 16; ++t) {
            p[t] = __expf(ev[t] - mnew);
            psum += p[t];
        }
        psum += __shfl_xor(psum, 16);
        const float scale = __expf(m - mnew);
        l = l * scale + psum;
        m = mnew;

        // ---- rescale accumulators per row (broadcast row scale via shfl)
#pragma unroll
        for (int r = 0; r < 8; ++r) {
            const int mloc = (lane < 16) ? r : (r + 8);
            const float sc = __shfl(scale, mloc);
#pragma unroll
            for (int nt = 0; nt < NT; ++nt) acc[nt][r] *= sc;
        }

        // ---- P fragment (16x32 bf16, A layout) and WMMAs against Wh chunk
        BF16x16 af;
#pragma unroll
        for (int t = 0; t < 16; ++t) af.v[t] = (__bf16)p[t];
#pragma unroll
        for (int nt = 0; nt < NT; ++nt) {
            const int col = nt * 16 + ncol;
            BF16x16 bf;
            bf.h[0] = *(const v8bf*)&sWhT[col * 40 + base];
            bf.h[1] = *(const v8bf*)&sWhT[col * 40 + 16 + base];
            acc[nt] = wmma_bf16(af.v, bf.v, acc[nt]);
        }
    }

    // ---- finalize: divide by denom, optional ELU, store
    const float inv = 1.0f / l;
#pragma unroll
    for (int r = 0; r < 8; ++r) {
        const int mloc = (lane < 16) ? r : (r + 8);
        const float ir = __shfl(inv, mloc);
#pragma unroll
        for (int nt = 0; nt < NT; ++nt) {
            const int col = nt * 16 + ncol;
            if (col < H) {
                float v = acc[nt][r] * ir;
                if (doElu) v = v > 0.0f ? v : (__expf(v) - 1.0f);
                out[((size_t)(b * Nn) + (i0 + mloc)) * H + col] = v;
            }
        }
    }
}

// ---------------------------------------------------------------------------
// Row-wise L2 normalization (D == blockDim.x == 128)
// ---------------------------------------------------------------------------
__global__ __launch_bounds__(128)
void l2norm_rows(const float* __restrict__ zin, float* __restrict__ zout, int D) {
    const int row = blockIdx.x;
    const int t = threadIdx.x;
    float v = zin[(size_t)row * D + t];
    float ss = v * v;
#pragma unroll
    for (int off = 16; off > 0; off >>= 1) ss += __shfl_down(ss, off);
    __shared__ float wsum[4];
    if ((t & 31) == 0) wsum[t >> 5] = ss;
    __syncthreads();
    float tot = wsum[0] + wsum[1] + wsum[2] + wsum[3];
    zout[(size_t)row * D + t] = v / sqrtf(tot);
}

// ---------------------------------------------------------------------------
// Host launcher
// ---------------------------------------------------------------------------
extern "C" void kernel_launch(void* const* d_in, const int* in_sizes, int n_in,
                              void* d_out, int out_size, void* d_ws, size_t ws_size,
                              hipStream_t stream) {
    (void)in_sizes; (void)n_in; (void)out_size; (void)ws_size;

    const int Bb = 64, Nn = 1024, FIN = 512, HID = 128, NCLS = 4;
    const int E = Nn * NCLS;             // 4096
    const int BN = Bb * Nn;              // 65536
    const int P1 = 1024, P2 = 128, C1 = 512, C2 = 2;

    const float* x          = (const float*)d_in[0];
    const float* adj        = (const float*)d_in[1];
    const float* W1         = (const float*)d_in[2];
    const float* a1         = (const float*)d_in[3];
    const float* W2         = (const float*)d_in[4];
    const float* a2         = (const float*)d_in[5];
    const float* in_proj_w  = (const float*)d_in[6];
    const float* in_proj_b  = (const float*)d_in[7];
    const float* out_proj_w = (const float*)d_in[8];
    const float* out_proj_b = (const float*)d_in[9];
    const float* p1_w       = (const float*)d_in[10];
    const float* p1_b       = (const float*)d_in[11];
    const float* p2_w       = (const float*)d_in[12];
    const float* p2_b       = (const float*)d_in[13];
    const float* c1_w       = (const float*)d_in[14];
    const float* c1_b       = (const float*)d_in[15];
    const float* c2_w       = (const float*)d_in[16];
    const float* c2_b       = (const float*)d_in[17];
    float* out = (float*)d_out;

    // workspace layout (floats)
    float* ws = (float*)d_ws;
    size_t o = 0;
    float* wh1   = ws + o; o += (size_t)BN * HID;   // 65536*128
    float* s1a   = ws + o; o += BN;
    float* s2a   = ws + o; o += BN;
    float* h1    = ws + o; o += (size_t)BN * HID;
    float* wh2   = ws + o; o += (size_t)BN * NCLS;
    float* s1b   = ws + o; o += BN;
    float* s2b   = ws + o; o += BN;
    float* flat2 = ws + o; o += (size_t)BN * NCLS;  // == [B, E]
    float* vbuf  = ws + o; o += (size_t)Bb * E;
    float* gbuf  = ws + o; o += (size_t)Bb * E;
    float* t1    = ws + o; o += (size_t)Bb * P1;
    float* zb    = ws + o; o += (size_t)Bb * P2;
    float* t2    = ws + o; o += (size_t)Bb * C1;

    auto cdiv = [](int a, int b) { return (a + b - 1) / b; };

    // --- GAT layer 1 ---
    gemm_bf16_wmma<<<dim3(cdiv(HID, 64), cdiv(BN, 128)), 256, 0, stream>>>(
        x, W1, nullptr, wh1, BN, FIN, HID, HID, 0);
    score_proj<<<cdiv(BN, 256), 256, 0, stream>>>(wh1, a1, s1a, s2a, BN, HID);
    gat_attention<8><<<Bb * (Nn / 64), 128, 0, stream>>>(
        wh1, adj, s1a, s2a, h1, Nn, HID, /*elu=*/1);

    // --- GAT layer 2 ---
    gemm_bf16_wmma<<<dim3(cdiv(NCLS, 64), cdiv(BN, 128)), 256, 0, stream>>>(
        h1, W2, nullptr, wh2, BN, HID, NCLS, NCLS, 0);
    score_proj<<<cdiv(BN, 256), 256, 0, stream>>>(wh2, a2, s1b, s2b, BN, NCLS);
    gat_attention<1><<<Bb * (Nn / 64), 128, 0, stream>>>(
        wh2, adj, s1b, s2b, flat2, Nn, NCLS, /*elu=*/0);

    // --- MHA with seq_len==1: softmax over one key == 1  =>  ctx = v ---
    // v = flat2 @ in_proj_w[:, 2E:3E] + in_proj_b[2E:]
    gemm_bf16_wmma<<<dim3(cdiv(E, 64), cdiv(Bb, 128)), 256, 0, stream>>>(
        flat2, in_proj_w + 2 * E, in_proj_b + 2 * E, vbuf, Bb, E, E, 3 * E, 0);
    // g = v @ out_proj_w + out_proj_b
    gemm_bf16_wmma<<<dim3(cdiv(E, 64), cdiv(Bb, 128)), 256, 0, stream>>>(
        vbuf, out_proj_w, out_proj_b, gbuf, Bb, E, E, E, 0);

    // --- contrastive head: z = normalize(relu(g@p1+b) @ p2 + b) ---
    gemm_bf16_wmma<<<dim3(cdiv(P1, 64), cdiv(Bb, 128)), 256, 0, stream>>>(
        gbuf, p1_w, p1_b, t1, Bb, E, P1, P1, 1);
    gemm_bf16_wmma<<<dim3(cdiv(P2, 64), cdiv(Bb, 128)), 256, 0, stream>>>(
        t1, p2_w, p2_b, zb, Bb, P1, P2, P2, 0);
    l2norm_rows<<<Bb, P2, 0, stream>>>(zb, out + Bb * C2, P2);

    // --- classifier: logits = relu(g@c1+b) @ c2 + b  (first in output tuple) ---
    gemm_bf16_wmma<<<dim3(cdiv(C1, 64), cdiv(Bb, 128)), 256, 0, stream>>>(
        gbuf, c1_w, c1_b, t2, Bb, E, C1, C1, 1);
    gemm_bf16_wmma<<<dim3(cdiv(C2, 64), cdiv(Bb, 128)), 256, 0, stream>>>(
        t2, c2_w, c2_b, out, Bb, C1, C2, C2, 0);
}
